// MLP_7189775254092
// MI455X (gfx1250) — 16-bit and 8-bit matrix operand order repaired and checked by reading, compiled, not run on hardware
//
#include <hip/hip_runtime.h>
#include <cstdint>
#include <cstddef>

typedef int      v8i  __attribute__((ext_vector_type(8)));
typedef float    v8f  __attribute__((ext_vector_type(8)));
typedef _Float16 v16h __attribute__((ext_vector_type(16)));
typedef _Float16 h8   __attribute__((ext_vector_type(8)));

#define BN_EPS 1e-5f

__device__ __forceinline__ void async_ld_b128(uint32_t ldsOff, const void* g) {
  asm volatile("global_load_async_to_lds_b128 %0, %1, off" :: "v"(ldsOff), "v"(g) : "memory");
}
__device__ __forceinline__ void wait_async0() {
  asm volatile("s_wait_asynccnt 0x0" ::: "memory");
}
__device__ __forceinline__ uint32_t lds_off(const void* p) {
  return (uint32_t)(uintptr_t)p;
}


__global__ void k_tern_w_f16(const float* __restrict__ w, _Float16* __restrict__ wq,
                             int N, int K, int Kp) {
  int idx = blockIdx.x * blockDim.x + threadIdx.x;
  if (idx >= N * Kp) return;
  int n = idx / Kp, k = idx - n * Kp;
  float v = (k < K) ? w[(size_t)n * K + k] : 0.0f;
  float q = fminf(1.0f, fmaxf(-1.0f, rintf(v)));
  wq[idx] = (_Float16)q;
}

__global__ void k_tern_w_i8(const float* __restrict__ w, int8_t* __restrict__ wq, int n) {
  int idx = blockIdx.x * blockDim.x + threadIdx.x;
  if (idx >= n) return;
  float q = fminf(1.0f, fmaxf(-1.0f, rintf(w[idx])));
  wq[idx] = (int8_t)q;
}

__global__ void k_cvt_x_f16(const float* __restrict__ x, _Float16* __restrict__ xh,
                            int M, int K, int Kp) {
  int idx = blockIdx.x * blockDim.x + threadIdx.x;
  if (idx >= M * Kp) return;
  int m = idx / Kp, k = idx - m * Kp;
  xh[idx] = (k < K) ? (_Float16)x[(size_t)m * K + k] : (_Float16)0.0f;
}

__global__ void k_zero_f32(float* __restrict__ p, int n) {
  int idx = blockIdx.x * blockDim.x + threadIdx.x;
  if (idx < n) p[idx] = 0.0f;
}

template <int M, int N, int Kp>
__global__ __launch_bounds__(256) void k_gemm_f16(const _Float16* __restrict__ A,
                                                  const _Float16* __restrict__ B,
                                                  float* __restrict__ C) {
  const int lane   = threadIdx.x & 31;
  const int wave   = threadIdx.x >> 5;
  const int lane16 = lane & 15;
  const int laneHi = lane >> 4;
  const int mBase  = blockIdx.x * 64  + (wave >> 2) * 32;
  const int nBase  = blockIdx.y * 128 + (wave & 3) * 32;

  v8f acc[2][2] = {};
  for (int k = 0; k < Kp; k += 32) {
    v16h a[2], b[2];
#pragma unroll
    for (int mt = 0; mt < 2; ++mt) {
      const _Float16* ap = A + (size_t)(mBase + mt * 16 + lane16) * Kp + k + 8 * laneHi;
      h8 lo = *(const h8*)(ap);
      h8 hi = *(const h8*)(ap + 16);
#pragma unroll
      for (int i = 0; i < 8; ++i) { a[mt][i] = lo[i]; a[mt][8 + i] = hi[i]; }
    }
#pragma unroll
    for (int nt = 0; nt < 2; ++nt) {
      const _Float16* bp = B + (size_t)(nBase + nt * 16 + lane16) * Kp + k + 8 * laneHi;
      h8 lo = *(const h8*)(bp);
      h8 hi = *(const h8*)(bp + 16);
#pragma unroll
      for (int i = 0; i < 8; ++i) { b[nt][i] = lo[i]; b[nt][8 + i] = hi[i]; }
    }
#pragma unroll
    for (int mt = 0; mt < 2; ++mt)
#pragma unroll
      for (int nt = 0; nt < 2; ++nt)
        acc[mt][nt] = __builtin_amdgcn_wmma_f32_16x16x32_f16(
            false, a[mt], false, b[nt], (short)0, acc[mt][nt], false, false);
  }
#pragma unroll
  for (int mt = 0; mt < 2; ++mt)
#pragma unroll
    for (int nt = 0; nt < 2; ++nt) {
      const int col  = nBase + nt * 16 + lane16;
      const int rowb = mBase + mt * 16 + 8 * laneHi;
#pragma unroll
      for (int r = 0; r < 8; ++r)
        C[(size_t)(rowb + r) * N + col] = acc[mt][nt][r];
    }
}

template <int M, int N, int K>
__global__ __launch_bounds__(256) void k_gemm_i8_lds(const int8_t* __restrict__ A,
                                                     const int8_t* __restrict__ B,
                                                     int* __restrict__ C) {
  constexpr int KS = 64;
  constexpr int NSTEP = K / KS;
  __shared__ __align__(16) int8_t shA[2][128][KS];
  __shared__ __align__(16) int8_t shB[2][128][KS];

  const int tid    = threadIdx.x;
  const int lane   = tid & 31;
  const int wave   = tid >> 5;
  const int lane16 = lane & 15;
  const int laneHi = lane >> 4;
  const int waveM  = wave >> 1;
  const int waveN  = wave & 1;
  const int blockM = blockIdx.x * 128;
  const int blockN = blockIdx.y * 128;

  auto stage = [&](int buf, int k) {
#pragma unroll
    for (int i = 0; i < 2; ++i) {
      const int c   = tid + i * 256;
      const int row = c >> 2;
      const int seg = (c & 3) * 16;
      async_ld_b128(lds_off(&shA[buf][row][seg]), A + (size_t)(blockM + row) * K + k + seg);
      async_ld_b128(lds_off(&shB[buf][row][seg]), B + (size_t)(blockN + row) * K + k + seg);
    }
  };

  v8i acc[2][4] = {};

  stage(0, 0);
  wait_async0();
  __syncthreads();

  for (int kt = 0; kt < NSTEP; ++kt) {
    const int buf = kt & 1;
    if (kt + 1 < NSTEP) stage(buf ^ 1, (kt + 1) * KS);

    v8i a[2], b[4];
#pragma unroll
    for (int mt = 0; mt < 2; ++mt) {
      const int8_t* ap = &shA[buf][waveM * 32 + mt * 16 + lane16][16 * laneHi];
      int2 p0 = *(const int2*)(ap);
      int2 p1 = *(const int2*)(ap + 8);
      int2 p2 = *(const int2*)(ap + 32);
      int2 p3 = *(const int2*)(ap + 40);
      a[mt][0] = p0.x; a[mt][1] = p0.y; a[mt][2] = p1.x; a[mt][3] = p1.y;
      a[mt][4] = p2.x; a[mt][5] = p2.y; a[mt][6] = p3.x; a[mt][7] = p3.y;
    }
#pragma unroll
    for (int nt = 0; nt < 4; ++nt) {
      const int8_t* bp = &shB[buf][waveN * 64 + nt * 16 + lane16][16 * laneHi];
      int4 q0 = *(const int4*)(bp);
      int4 q1 = *(const int4*)(bp + 32);
      b[nt][0] = q0.x; b[nt][1] = q0.y; b[nt][2] = q0.z; b[nt][3] = q0.w;
      b[nt][4] = q1.x; b[nt][5] = q1.y; b[nt][6] = q1.z; b[nt][7] = q1.w;
    }
#pragma unroll
    for (int mt = 0; mt < 2; ++mt)
#pragma unroll
      for (int nt = 0; nt < 4; ++nt)
        acc[mt][nt] = __builtin_amdgcn_wmma_i32_16x16x64_iu8(
            true, a[mt], true, b[nt], acc[mt][nt], false, false);

    wait_async0();
    __syncthreads();
  }

#pragma unroll
  for (int mt = 0; mt < 2; ++mt)
#pragma unroll
    for (int nt = 0; nt < 4; ++nt) {
      const int col  = blockN + waveN * 64 + nt * 16 + lane16;
      const int rowb = blockM + waveM * 32 + mt * 16 + 8 * laneHi;
#pragma unroll
      for (int r = 0; r < 8; ++r)
        C[(size_t)(rowb + r) * N + col] = acc[mt][nt][r];
    }
}

template <int N, int K>
__global__ void k_gemm4_i8(const int8_t* __restrict__ A, const int8_t* __restrict__ W,
                           float* __restrict__ Y, int M) {
  int idx = blockIdx.x * blockDim.x + threadIdx.x;
  if (idx >= M * N) return;
  int m = idx / N, n = idx - m * N;
  const char4* ar = (const char4*)(A + (size_t)m * K);
  const char4* wr = (const char4*)(W + (size_t)n * K);
  int s = 0;
  for (int k = 0; k < K / 4; ++k) {
    char4 a = ar[k], w = wr[k];
    s += (int)a.x * (int)w.x + (int)a.y * (int)w.y + (int)a.z * (int)w.z + (int)a.w * (int)w.w;
  }
  Y[idx] = (float)s;
}


__global__ void k_colstats_f32(const float* __restrict__ y, float* __restrict__ sum,
                               float* __restrict__ sumsq, int M, int N, int rpc) {
  int col = blockIdx.x * blockDim.x + threadIdx.x;
  if (col >= N) return;
  int r0 = blockIdx.y * rpc;
  int r1 = min(M, r0 + rpc);
  float s = 0.0f, s2 = 0.0f;
  for (int r = r0; r < r1; ++r) {
    float v = y[(size_t)r * N + col];
    s += v; s2 += v * v;
  }
  atomicAdd(&sum[col], s);
  atomicAdd(&sumsq[col], s2);
}

__global__ void k_colstats_i32(const int* __restrict__ y, float* __restrict__ sum,
                               float* __restrict__ sumsq, int M, int N, int rpc) {
  int col = blockIdx.x * blockDim.x + threadIdx.x;
  if (col >= N) return;
  int r0 = blockIdx.y * rpc;
  int r1 = min(M, r0 + rpc);
  float s = 0.0f, s2 = 0.0f;
  for (int r = r0; r < r1; ++r) {
    float v = (float)y[(size_t)r * N + col];
    s += v; s2 += v * v;
  }
  atomicAdd(&sum[col], s);
  atomicAdd(&sumsq[col], s2);
}

__global__ void k_bn_quant_f32(const float* __restrict__ y, const float* __restrict__ sum,
                               const float* __restrict__ sumsq, const float* __restrict__ g,
                               const float* __restrict__ b, int8_t* __restrict__ h,
                               int M, int N) {
  int idx = blockIdx.x * blockDim.x + threadIdx.x;
  if (idx >= M * N) return;
  int col = idx % N;
  float invM  = 1.0f / (float)M;
  float mean  = sum[col] * invM;
  float var   = sumsq[col] * invM - mean * mean;
  float sc    = g[col] * rsqrtf(var + BN_EPS);
  float v     = (y[idx] - mean) * sc + b[col];
  float q     = fminf(1.0f, fmaxf(-1.0f, rintf(v)));
  h[idx] = (int8_t)q;
}

__global__ void k_bn_quant_i32(const int* __restrict__ y, const float* __restrict__ sum,
                               const float* __restrict__ sumsq, const float* __restrict__ g,
                               const float* __restrict__ b, int8_t* __restrict__ h,
                               int M, int N) {
  int idx = blockIdx.x * blockDim.x + threadIdx.x;
  if (idx >= M * N) return;
  int col = idx % N;
  float invM  = 1.0f / (float)M;
  float mean  = sum[col] * invM;
  float var   = sumsq[col] * invM - mean * mean;
  float sc    = g[col] * rsqrtf(var + BN_EPS);
  float v     = ((float)y[idx] - mean) * sc + b[col];
  float q     = fminf(1.0f, fmaxf(-1.0f, rintf(v)));
  h[idx] = (int8_t)q;
}

__global__ void k_bn_f32(const float* __restrict__ y, const float* __restrict__ sum,
                         const float* __restrict__ sumsq, const float* __restrict__ g,
                         const float* __restrict__ b, float* __restrict__ out,
                         int M, int N) {
  int idx = blockIdx.x * blockDim.x + threadIdx.x;
  if (idx >= M * N) return;
  int col = idx % N;
  float invM  = 1.0f / (float)M;
  float mean  = sum[col] * invM;
  float var   = sumsq[col] * invM - mean * mean;
  float sc    = g[col] * rsqrtf(var + BN_EPS);
  out[idx] = (y[idx] - mean) * sc + b[col];
}


extern "C" void kernel_launch(void* const* d_in, const int* in_sizes, int n_in,
                              void* d_out, int out_size, void* d_ws, size_t ws_size,
                              hipStream_t stream) {
  (void)in_sizes; (void)n_in; (void)out_size; (void)ws_size;
  constexpr int Bm = 8192, D = 784, NH = 2048, Cc = 10, Kp1 = 800;

  const float* x  = (const float*)d_in[0];
  const float* w1 = (const float*)d_in[1];
  const float* g1 = (const float*)d_in[2];
  const float* b1 = (const float*)d_in[3];
  const float* w2 = (const float*)d_in[4];
  const float* g2 = (const float*)d_in[5];
  const float* b2 = (const float*)d_in[6];
  const float* w3 = (const float*)d_in[7];
  const float* g3 = (const float*)d_in[8];
  const float* b3 = (const float*)d_in[9];
  const float* w4 = (const float*)d_in[10];
  const float* g4 = (const float*)d_in[11];
  const float* b4 = (const float*)d_in[12];
  float* out = (float*)d_out;

  char* ws = (char*)d_ws;
  _Float16* xh    = (_Float16*)(ws + 0);
  _Float16* w1q   = (_Float16*)(ws + 13107200);
  int8_t*   w2q   = (int8_t*)  (ws + 16384000);
  int8_t*   w3q   = (int8_t*)  (ws + 20578304);
  int8_t*   w4q   = (int8_t*)  (ws + 24772608);
  float*    yf    = (float*)   (ws + 24797184);
  int*      yi    = (int*)     (ws + 24797184);
  int8_t*   h     = (int8_t*)  (ws + 91906048);
  float*    y4    = (float*)   (ws + 108683264);
  float*    sum   = (float*)   (ws + 109010944);
  float*    sumsq = (float*)   (ws + 109019136);

  const int T = 256;
  dim3 gemm1Grid(Bm / 64, NH / 128);
  dim3 gemmIGrid(Bm / 128, NH / 128);
  dim3 statGrid((NH + T - 1) / T, 32);
  dim3 statGrid4(1, 32);
  int rpc = Bm / 32;

  k_tern_w_f16<<<(NH * Kp1 + T - 1) / T, T, 0, stream>>>(w1, w1q, NH, D, Kp1);
  k_tern_w_i8 <<<(NH * NH + T - 1) / T, T, 0, stream>>>(w2, w2q, NH * NH);
  k_tern_w_i8 <<<(NH * NH + T - 1) / T, T, 0, stream>>>(w3, w3q, NH * NH);
  k_tern_w_i8 <<<(Cc * NH + T - 1) / T, T, 0, stream>>>(w4, w4q, Cc * NH);
  k_cvt_x_f16 <<<(Bm * Kp1 + T - 1) / T, T, 0, stream>>>(x, xh, Bm, D, Kp1);

  k_gemm_f16<Bm, NH, Kp1><<<gemm1Grid, T, 0, stream>>>(xh, w1q, yf);
  k_zero_f32<<<(2 * NH + T - 1) / T, T, 0, stream>>>(sum, 2 * NH);
  k_colstats_f32<<<statGrid, T, 0, stream>>>(yf, sum, sumsq, Bm, NH, rpc);
  k_bn_quant_f32<<<(Bm * NH + T - 1) / T, T, 0, stream>>>(yf, sum, sumsq, g1, b1, h, Bm, NH);

  k_gemm_i8_lds<Bm, NH, NH><<<gemmIGrid, T, 0, stream>>>(h, w2q, yi);
  k_zero_f32<<<(2 * NH + T - 1) / T, T, 0, stream>>>(sum, 2 * NH);
  k_colstats_i32<<<statGrid, T, 0, stream>>>(yi, sum, sumsq, Bm, NH, rpc);
  k_bn_quant_i32<<<(Bm * NH + T - 1) / T, T, 0, stream>>>(yi, sum, sumsq, g2, b2, h, Bm, NH);

  k_gemm_i8_lds<Bm, NH, NH><<<gemmIGrid, T, 0, stream>>>(h, w3q, yi);
  k_zero_f32<<<(2 * NH + T - 1) / T, T, 0, stream>>>(sum, 2 * NH);
  k_colstats_i32<<<statGrid, T, 0, stream>>>(yi, sum, sumsq, Bm, NH, rpc);
  k_bn_quant_i32<<<(Bm * NH + T - 1) / T, T, 0, stream>>>(yi, sum, sumsq, g3, b3, h, Bm, NH);

  k_gemm4_i8<Cc, NH><<<(Bm * Cc + T - 1) / T, T, 0, stream>>>(h, w4q, y4, Bm);
  k_zero_f32<<<(2 * NH + T - 1) / T, T, 0, stream>>>(sum, 2 * NH);
  k_colstats_f32<<<statGrid4, T, 0, stream>>>(y4, sum, sumsq, Bm, Cc, rpc);
  k_bn_f32<<<(Bm * Cc + T - 1) / T, T, 0, stream>>>(y4, sum, sumsq, g4, b4, out, Bm, Cc);
}
